// VectorQuantizedEmbeddings_26319559590100
// MI455X (gfx1250) — matrix operand order repaired and checked by reading, not compiled, not run on hardware
//
#include <hip/hip_runtime.h>
#include <hip/hip_bf16.h>

// ---------------------------------------------------------------------------
// VQ embedding lookup for MI455X (gfx1250, wave32, WMMA).
//   z:         [16, 4096, 256] f32  (N = 65536 tokens, D = 256)
//   embedding: [2048, 256]     f32  (K = 2048 codes)
// Outputs (concatenated in d_out, f32):
//   quantized [N, 256], token_ids [N] (written as float values)
//
// Main kernel: 256 blocks x 8 waves; each wave owns 32 tokens (two 16-row
// A slabs in registers). The block cooperatively double-buffers 16-code
// B tiles (bf16) through LDS so the 1 MB codebook is read from L2 once per
// block (not once per wave), and each B fragment feeds two WMMA chains.
// ---------------------------------------------------------------------------

typedef __bf16 bf16_t;
typedef __attribute__((ext_vector_type(16))) bf16_t v16bf;
typedef __attribute__((ext_vector_type(8)))  bf16_t v8bf;
typedef __attribute__((ext_vector_type(8)))  float  v8f;
typedef __attribute__((ext_vector_type(4)))  float  v4f;

#define VQ_N 65536
#define VQ_K 2048
#define VQ_D 256
#define NTILES (VQ_K / 16)     // 128 code tiles
#define LDS_STRIDE 264         // 256 + 8 pad: 4-bank rotation/row, 16B aligned

// ---------------------------------------------------------------------------
// Prep: codebook f32 -> bf16 (WMMA B operand) + exact f32 e_sq[k].
// ---------------------------------------------------------------------------
__global__ __launch_bounds__(256) void vq_prep(const float* __restrict__ E,
                                               bf16_t* __restrict__ Ebf,
                                               float* __restrict__ esq) {
    const int k = blockIdx.x;
    const int d = threadIdx.x;
    const float v = E[(size_t)k * VQ_D + d];
    Ebf[(size_t)k * VQ_D + d] = (bf16_t)v;

    float s = v * v;
    #pragma unroll
    for (int m = 16; m >= 1; m >>= 1) s += __shfl_xor(s, m, 32);

    __shared__ float red[8];
    if ((threadIdx.x & 31) == 0) red[threadIdx.x >> 5] = s;
    __syncthreads();
    if (threadIdx.x == 0) {
        float t = 0.0f;
        #pragma unroll
        for (int i = 0; i < 8; ++i) t += red[i];
        esq[k] = t;
    }
}

// ---------------------------------------------------------------------------
// Main kernel.
// ---------------------------------------------------------------------------
__global__ __launch_bounds__(256) void vq_main(const float* __restrict__ z,
                                               const float* __restrict__ E,
                                               const bf16_t* __restrict__ Ebf,
                                               const float* __restrict__ esq,
                                               float* __restrict__ outq,
                                               float* __restrict__ outid) {
    __shared__ bf16_t ldsb[2][16 * LDS_STRIDE];   // double-buffered B tiles

    const int lane = threadIdx.x & 31;
    const int wave = threadIdx.x >> 5;
    const int row  = lane & 15;       // A row / B column within 16-tile
    const int half = lane >> 4;       // 0: lanes 0-15, 1: lanes 16-31
    const int token_base = (blockIdx.x * 8 + wave) * 32;

    // ---- Load two A slabs (32 tokens x 256 D) as bf16, ISA A-layout ----
    // lanes 0-15:  elems 0-7 -> d=c*32+0..7,  elems 8-15 -> d=c*32+16..23
    // lanes 16-31: elems 0-7 -> d=c*32+8..15, elems 8-15 -> d=c*32+24..31
    const int kbaseA = half * 8;
    v16bf a0[8], a1[8];
    {
        const float* zr0 = z + (size_t)(token_base + row) * VQ_D;
        const float* zr1 = z + (size_t)(token_base + 16 + row) * VQ_D;
        #pragma unroll
        for (int c = 0; c < 8; ++c) {
            const int d0 = c * 32 + kbaseA;
            #pragma unroll
            for (int s = 0; s < 2; ++s) {
                const float* zr = s ? zr1 : zr0;
                v4f p0 = *(const v4f*)(zr + d0);
                v4f p1 = *(const v4f*)(zr + d0 + 4);
                v4f p2 = *(const v4f*)(zr + d0 + 16);
                v4f p3 = *(const v4f*)(zr + d0 + 20);
                v16bf t;
                #pragma unroll
                for (int j = 0; j < 4; ++j) {
                    t[j]      = (bf16_t)p0[j];
                    t[4 + j]  = (bf16_t)p1[j];
                    t[8 + j]  = (bf16_t)p2[j];
                    t[12 + j] = (bf16_t)p3[j];
                }
                if (s) a1[c] = t; else a0[c] = t;
            }
        }
    }

    float minv0[8], minv1[8];
    int   mini0[8], mini1[8];
    #pragma unroll
    for (int r = 0; r < 8; ++r) {
        minv0[r] = minv1[r] = 3.4028235e38f;
        mini0[r] = mini1[r] = 0;
    }

    // ---- Stage tile 0 into LDS buffer 0 (coalesced: tile is contiguous) ----
    #pragma unroll
    for (int i = 0; i < 2; ++i) {
        const int g   = threadIdx.x * 8 + i * 2048;   // elem within 16x256 tile
        const int r   = g >> 8;
        const int col = g & 255;
        *(v8bf*)(&ldsb[0][r * LDS_STRIDE + col]) = *(const v8bf*)(Ebf + g);
    }
    __syncthreads();

    // ---- Sweep all K=2048 codes; double-buffered B tiles in LDS ----
    // B layout (16-bit, 32x16): lanes 0-15 K=0-7,16-23, lanes 16-31 K=8-15,24-31,
    // column N = lane&15 -> two 16B ds reads per k-chunk.
    const int kbaseB = half * 8;
    for (int ct = 0; ct < NTILES; ++ct) {
        const int cur = ct & 1;

        if (ct + 1 < NTILES) {   // stage next tile while computing this one
            #pragma unroll
            for (int i = 0; i < 2; ++i) {
                const int g   = threadIdx.x * 8 + i * 2048;
                const int r   = g >> 8;
                const int col = g & 255;
                *(v8bf*)(&ldsb[cur ^ 1][r * LDS_STRIDE + col]) =
                    *(const v8bf*)(Ebf + (size_t)(ct + 1) * 4096 + g);
            }
        }

        const bf16_t* bbase = &ldsb[cur][row * LDS_STRIDE + kbaseB];
        v8f acc0 = {};
        v8f acc1 = {};
        #pragma unroll
        for (int c = 0; c < 8; ++c) {
            v16bf b = __builtin_shufflevector(*(const v8bf*)(bbase + c * 32), *(const v8bf*)(bbase + c * 32 + 16), 0, 1, 2, 3, 4, 5, 6, 7, 8, 9, 10, 11, 12, 13, 14, 15);
            acc0 = __builtin_amdgcn_wmma_f32_16x16x32_bf16(
                false, a0[c], false, b, (short)0, acc0, false, false);
            acc1 = __builtin_amdgcn_wmma_f32_16x16x32_bf16(
                false, a1[c], false, b, (short)0, acc1, false, false);
        }

        const float es = esq[ct * 16 + row];   // exact f32 e_sq of this column
        const int   ci = ct * 16 + row;
        #pragma unroll
        for (int r = 0; r < 8; ++r) {
            const float s0 = fmaf(-2.0f, acc0[r], es);
            const float s1 = fmaf(-2.0f, acc1[r], es);
            if (s0 < minv0[r]) { minv0[r] = s0; mini0[r] = ci; }
            if (s1 < minv1[r]) { minv1[r] = s1; mini1[r] = ci; }
        }
        __syncthreads();   // staging of next buf done + reads of cur done
    }

    // ---- Cross-lane argmin within each 16-lane group, then write out ----
    #pragma unroll
    for (int s = 0; s < 2; ++s) {
        float* minv = s ? minv1 : minv0;
        int*   mini = s ? mini1 : mini0;
        #pragma unroll
        for (int r = 0; r < 8; ++r) {
            float v = minv[r];
            int   i = mini[r];
            #pragma unroll
            for (int m = 8; m >= 1; m >>= 1) {
                const float ov = __shfl_xor(v, m, 32);
                const int   oi = __shfl_xor(i, m, 32);
                if (ov < v || (ov == v && oi < i)) { v = ov; i = oi; }
            }
            // C layout: VGPR r -> row M=r (lanes 0-15) / M=r+8 (lanes 16-31)
            const int token = token_base + s * 16 + r + half * 8;
            if (row == 0) outid[token] = (float)i;
            const float* srow = E + (size_t)i * VQ_D;    // exact f32 row
            float* orow = outq + (size_t)token * VQ_D;
            #pragma unroll
            for (int q = 0; q < 4; ++q) {                // coalesced 128b I/O
                *(v4f*)(orow + (q * 16 + row) * 4) =
                    *(const v4f*)(srow + (q * 16 + row) * 4);
            }
        }
    }
}

extern "C" void kernel_launch(void* const* d_in, const int* in_sizes, int n_in,
                              void* d_out, int out_size, void* d_ws, size_t ws_size,
                              hipStream_t stream) {
    const float* z = (const float*)d_in[0];   // [16,4096,256] f32
    const float* E = (const float*)d_in[1];   // [2048,256]    f32

    bf16_t* Ebf = (bf16_t*)d_ws;                                   // 1 MB
    float*  esq = (float*)((char*)d_ws + (size_t)VQ_K * VQ_D * 2); // 8 KB

    float* outq  = (float*)d_out;                 // [N, 256] quantized
    float* outid = outq + (size_t)VQ_N * VQ_D;    // [N] token ids (as float)

    vq_prep<<<VQ_K, 256, 0, stream>>>(E, Ebf, esq);
    // 8 waves/block, 32 tokens/wave -> 256 tokens/block
    vq_main<<<VQ_N / 256, 256, 0, stream>>>(z, E, Ebf, esq, outq, outid);
}
